// BDH_GPU_4853313044969
// MI455X (gfx1250) — matrix operand order repaired and checked by reading, not compiled, not run on hardware
//
#include <hip/hip_runtime.h>

// ---------------------------------------------------------------------------
// BDH forward on gfx1250 (MI455X).
//  - bf16 WMMA (v_wmma_f32_16x16x32_bf16), f32 accumulate, 2x2 register tiles
//  - A panels staged to LDS by the Tensor Data Mover (TENSOR_LOAD_TO_LDS),
//    double-buffered, with D# padding for conflict-free ds_load_b128 reads
//  - deterministic split-K for the K=32768 encoder GEMM
// ---------------------------------------------------------------------------

#define Dm    256
#define Hh    4
#define Lnum  6
#define Ntot  32768
#define NHd   8192
#define Tt    1024
#define VOC   256
#define HALF  4096
#define KS    8          // split-K factor for encoder GEMM
#define LPITCH 264       // LDS row pitch in bf16 elems: 256 data + 8 pad (528 B)

typedef __bf16 bh8   __attribute__((ext_vector_type(8)));
typedef __bf16 bh16  __attribute__((ext_vector_type(16)));
typedef float  f32x8 __attribute__((ext_vector_type(8)));
typedef unsigned int u32x4 __attribute__((ext_vector_type(4)));
typedef int          i32x4 __attribute__((ext_vector_type(4)));
typedef int          i32x8 __attribute__((ext_vector_type(8)));

// ------------------------------ helpers ------------------------------------

__device__ __forceinline__ float wave_sum(float x) {
  #pragma unroll
  for (int off = 16; off > 0; off >>= 1) x += __shfl_xor(x, off, 32);
  return x;
}

// TDM: DMA a 32-row x 256-elem bf16 tile (row stride = stride_elems) from
// global into LDS at lds_addr, inserting 16B padding after each 512B row so
// the LDS pitch is 528B (LPITCH).  2D descriptor, groups 2/3/iter zero.
__device__ __forceinline__ void tdm_load_panel(const __bf16* gsrc,
                                               unsigned lds_addr,
                                               long long stride_elems) {
  unsigned long long ga = (unsigned long long)(uintptr_t)gsrc;
  u32x4 g0;
  g0[0] = 1u;                                   // count=1, user mode
  g0[1] = lds_addr;                             // LDS byte address
  g0[2] = (unsigned)ga;                         // global_addr[31:0]
  g0[3] = (unsigned)((ga >> 32) & 0x01FFFFFFu) | (2u << 30);  // [56:32]|type=2
  const unsigned td0 = 1u << 20, td1 = 1u << 20;      // generous OOB dims
  unsigned long long s0 = (unsigned long long)stride_elems;
  i32x8 g1;
  // data_size=1(2B) | pad_enable | pad_interval=6(128 dw) | pad_amount=3(4 dw)
  g1[0] = (int)((1u << 16) | (1u << 20) | (6u << 22) | (3u << 25));
  g1[1] = (int)((td0 & 0xFFFFu) << 16);               // [63:48]=td0 lo
  g1[2] = (int)(((td0 >> 16) & 0xFFFFu) | ((td1 & 0xFFFFu) << 16));
  g1[3] = (int)(((td1 >> 16) & 0xFFFFu) | (256u << 16));   // tile_dim0=256
  g1[4] = (int)(32u);                                 // tile_dim1=32, dim2=0
  g1[5] = (int)(unsigned)(s0 & 0xFFFFFFFFu);          // dim0_stride[31:0]
  g1[6] = (int)(unsigned)((s0 >> 32) & 0xFFFFu);      // dim0_stride[47:32]
  g1[7] = 0;
  i32x4 gz4 = {0, 0, 0, 0};
  i32x8 gz8 = {0, 0, 0, 0, 0, 0, 0, 0};
  __builtin_amdgcn_tensor_load_to_lds(g0, g1, gz4, gz4, gz8, 0);
}

// A fragment (16x32 bf16) from an LDS panel with LPITCH row stride.
__device__ __forceinline__ bh16 fragA_lds(const __bf16* sm, int row0, int k0) {
  int lane = threadIdx.x & 31;
  int r    = row0 + (lane & 15);
  int koff = (lane >> 4) * 8;
  const __bf16* p = sm + r * LPITCH + k0 + koff;
  bh8 a0 = *reinterpret_cast<const bh8*>(p);
  bh8 a1 = *reinterpret_cast<const bh8*>(p + 16);
  bh16 f;
  #pragma unroll
  for (int i = 0; i < 8; ++i) { f[i] = a0[i]; f[8 + i] = a1[i]; }
  return f;
}

// A fragment from global row-major source.
__device__ __forceinline__ bh16 load_fragA(const __bf16* base, int lda,
                                           int row0, int k0) {
  int lane = threadIdx.x & 31;
  int r    = row0 + (lane & 15);
  int koff = (lane >> 4) * 8;
  const __bf16* p = base + (size_t)r * lda + k0 + koff;
  bh8 a0 = *reinterpret_cast<const bh8*>(p);
  bh8 a1 = *reinterpret_cast<const bh8*>(p + 16);
  bh16 f;
  #pragma unroll
  for (int i = 0; i < 8; ++i) { f[i] = a0[i]; f[8 + i] = a1[i]; }
  return f;
}

// B fragment (32x16) from an N-major (transposed) buffer BT: BT[n][k]=B[k][n].
__device__ __forceinline__ bh16 load_fragB(const __bf16* baseT, int ldb,
                                           int col0, int k0) {
  int lane = threadIdx.x & 31;
  int c    = col0 + (lane & 15);
  int koff = (lane >> 4) * 8;
  const __bf16* p = baseT + (size_t)c * ldb + k0 + koff;
  bh8 b0 = *reinterpret_cast<const bh8*>(p);
  bh8 b1 = *reinterpret_cast<const bh8*>(p + 16);
  bh16 f;
  #pragma unroll
  for (int i = 0; i < 8; ++i) { f[i] = b0[i]; f[8 + i] = b1[i]; }
  return f;
}

__device__ __forceinline__ f32x8 wmma_bf16(bh16 a, bh16 b, f32x8 c) {
  return __builtin_amdgcn_wmma_f32_16x16x32_bf16(
      false, a, false, b, (short)0, c, false, false);
}

// --------------------------- weight transpose ------------------------------
__global__ void k_cvt_T(const float* __restrict__ src, __bf16* __restrict__ dst,
                        int R, int C) {
  __shared__ __bf16 tile[32][33];
  int bx = blockIdx.x * 32, by = blockIdx.y * 32;
  int tx = threadIdx.x, ty = threadIdx.y;
  #pragma unroll
  for (int j = 0; j < 32; j += 8)
    tile[ty + j][tx] = (__bf16)src[(size_t)(by + ty + j) * C + (bx + tx)];
  __syncthreads();
  #pragma unroll
  for (int j = 0; j < 32; j += 8)
    dst[(size_t)(bx + ty + j) * R + (by + tx)] = tile[tx][ty + j];
}

// ------------------------- embedding + layernorm ---------------------------
__global__ void k_embed(const int* __restrict__ idx,
                        const float* __restrict__ wte,
                        float* __restrict__ v, __bf16* __restrict__ vb,
                        __bf16* __restrict__ vbT) {
  int row = blockIdx.x * 8 + (threadIdx.x >> 5), lane = threadIdx.x & 31;
  if (row >= Tt) return;
  const float* w = wte + (size_t)idx[row] * Dm;
  float x[8];
  #pragma unroll
  for (int i = 0; i < 8; ++i) x[i] = w[lane + 32 * i];
  float s = 0.f, s2 = 0.f;
  #pragma unroll
  for (int i = 0; i < 8; ++i) { s += x[i]; s2 += x[i] * x[i]; }
  float m  = wave_sum(s) * (1.f / Dm);
  float vv = wave_sum(s2) * (1.f / Dm) - m * m;
  float r  = rsqrtf(vv + 1e-5f);
  #pragma unroll
  for (int i = 0; i < 8; ++i) {
    int c = lane + 32 * i;
    float y = (x[i] - m) * r;
    v[(size_t)row * Dm + c]   = y;
    vb[(size_t)row * Dm + c]  = (__bf16)y;
    vbT[(size_t)c * Tt + row] = (__bf16)y;
  }
}

// ------------- GEMM1: x = relu(v @ decoder_x[h]) ; TDM-staged A ------------
__global__ void k_gemm_x(const __bf16* __restrict__ A,    // v bf16 1024x256
                         const __bf16* __restrict__ BT,   // decxT H x 8192x256
                         __bf16* __restrict__ X) {        // H x 1024x8192
  __shared__ __bf16 sA[32 * LPITCH];
  int h = blockIdx.z, t0 = blockIdx.y * 32;
  int w = threadIdx.x >> 5, lane = threadIdx.x & 31;
  int n0 = blockIdx.x * 256 + w * 32;
  if (threadIdx.x < 32)
    tdm_load_panel(A + (size_t)t0 * Dm, (unsigned)(uintptr_t)sA, Dm);
  __builtin_amdgcn_s_wait_tensorcnt(0);
  __syncthreads();
  const __bf16* Bh = BT + (size_t)h * NHd * Dm;
  f32x8 c00 = {}, c01 = {}, c10 = {}, c11 = {};
  #pragma unroll
  for (int k0 = 0; k0 < Dm; k0 += 32) {
    bh16 a0 = fragA_lds(sA, 0, k0),  a1 = fragA_lds(sA, 16, k0);
    bh16 b0 = load_fragB(Bh, Dm, n0, k0), b1 = load_fragB(Bh, Dm, n0 + 16, k0);
    c00 = wmma_bf16(a0, b0, c00); c01 = wmma_bf16(a0, b1, c01);
    c10 = wmma_bf16(a1, b0, c10); c11 = wmma_bf16(a1, b1, c11);
  }
  int col = n0 + (lane & 15), rb = (lane >> 4) * 8;
  __bf16* Xh = X + (size_t)h * Tt * NHd;
  #pragma unroll
  for (int j = 0; j < 8; ++j) {
    int r = t0 + rb + j;
    Xh[(size_t)r * NHd + col]             = (__bf16)fmaxf(c00[j], 0.f);
    Xh[(size_t)r * NHd + col + 16]        = (__bf16)fmaxf(c01[j], 0.f);
    Xh[(size_t)(r + 16) * NHd + col]      = (__bf16)fmaxf(c10[j], 0.f);
    Xh[(size_t)(r + 16) * NHd + col + 16] = (__bf16)fmaxf(c11[j], 0.f);
  }
}

// ------------------------------- rope --------------------------------------
__global__ void k_rope(const __bf16* __restrict__ X, __bf16* __restrict__ XR) {
  long long i = (long long)blockIdx.x * blockDim.x + threadIdx.x;
  const long long total = (long long)Hh * Tt * HALF;
  if (i >= total) return;
  int f = (int)(i % HALF);
  int t = (int)((i / HALF) % Tt);
  int h = (int)(i / ((long long)Tt * HALF));
  size_t base = ((size_t)h * Tt + t) * NHd;
  float x1 = (float)X[base + f], x2 = (float)X[base + f + HALF];
  float inv = __expf(-(float)f * (9.2103403719761836f / (float)HALF));
  float ang = (float)t * inv;
  float sn, cs; __sincosf(ang, &sn, &cs);
  XR[base + f]        = (__bf16)(x1 * cs - x2 * sn);
  XR[base + f + HALF] = (__bf16)(x2 * cs + x1 * sn);
}

// -------- GEMM2: scores = causal(xr @ xr^T), K=8192, TDM double-buffer -----
__global__ void k_gemm_scores(const __bf16* __restrict__ XR,
                              __bf16* __restrict__ S) {    // H x 1024x1024
  __shared__ __bf16 sA[2][32 * LPITCH];
  int h = blockIdx.z, t0 = blockIdx.y * 32, sblk = blockIdx.x * 256;
  if (sblk > t0 + 32) return;               // whole block above diagonal band
  int w = threadIdx.x >> 5, lane = threadIdx.x & 31;
  int s0 = sblk + w * 32;
  bool active = (s0 <= t0 + 32);
  bool wave0  = (threadIdx.x < 32);
  const __bf16* Xh = XR + (size_t)h * Tt * NHd;
  const int P = NHd / 256;                  // 32 K-panels of 256
  if (wave0)
    tdm_load_panel(Xh + (size_t)t0 * NHd, (unsigned)(uintptr_t)sA[0], NHd);
  f32x8 c00 = {}, c01 = {}, c10 = {}, c11 = {};
  for (int p = 0; p < P; ++p) {
    if (wave0 && p + 1 < P)
      tdm_load_panel(Xh + (size_t)t0 * NHd + (p + 1) * 256,
                     (unsigned)(uintptr_t)sA[(p + 1) & 1], NHd);
    if (p + 1 < P) __builtin_amdgcn_s_wait_tensorcnt(1);
    else           __builtin_amdgcn_s_wait_tensorcnt(0);
    __syncthreads();
    if (active) {
      const __bf16* sm = sA[p & 1];
      int kp = p * 256;
      #pragma unroll
      for (int k0 = 0; k0 < 256; k0 += 32) {
        __builtin_prefetch(Xh + (size_t)(s0 + (lane & 15)) * NHd + kp + k0 + 256, 0, 3);
        bh16 a0 = fragA_lds(sm, 0, k0),  a1 = fragA_lds(sm, 16, k0);
        bh16 b0 = load_fragB(Xh, NHd, s0, kp + k0);
        bh16 b1 = load_fragB(Xh, NHd, s0 + 16, kp + k0);
        c00 = wmma_bf16(a0, b0, c00); c01 = wmma_bf16(a0, b1, c01);
        c10 = wmma_bf16(a1, b0, c10); c11 = wmma_bf16(a1, b1, c11);
      }
    }
    __syncthreads();
  }
  if (!active) return;
  int col = s0 + (lane & 15), rb = (lane >> 4) * 8;
  __bf16* Sh = S + (size_t)h * Tt * Tt;
  #pragma unroll
  for (int j = 0; j < 8; ++j) {
    int r = t0 + rb + j;
    Sh[(size_t)r * Tt + col]             = (__bf16)((col      < r     ) ? c00[j] : 0.f);
    Sh[(size_t)r * Tt + col + 16]        = (__bf16)((col + 16 < r     ) ? c01[j] : 0.f);
    Sh[(size_t)(r + 16) * Tt + col]      = (__bf16)((col      < r + 16) ? c10[j] : 0.f);
    Sh[(size_t)(r + 16) * Tt + col + 16] = (__bf16)((col + 16 < r + 16) ? c11[j] : 0.f);
  }
}

// --------------- GEMM3: a = scores @ v   (K truncated at diag) -------------
__global__ void k_gemm_av(const __bf16* __restrict__ S,
                          const __bf16* __restrict__ vT,   // 256 x 1024
                          float* __restrict__ Aout) {      // H x 1024x256
  int h = blockIdx.z, t0 = blockIdx.y * 16;
  int n0 = (blockIdx.x * 8 + (threadIdx.x >> 5)) * 16;
  const __bf16* Sh = S + (size_t)h * Tt * Tt;
  f32x8 acc = {};
  for (int k0 = 0; k0 < t0 + 16; k0 += 32)
    acc = wmma_bf16(load_fragA(Sh, Tt, t0, k0),
                    load_fragB(vT, Tt, n0, k0), acc);
  int lane = threadIdx.x & 31, col = n0 + (lane & 15), rb = (lane >> 4) * 8;
  float* Ah = Aout + (size_t)h * Tt * Dm;
  #pragma unroll
  for (int j = 0; j < 8; ++j)
    Ah[(size_t)(t0 + rb + j) * Dm + col] = acc[j];
}

// ----------------------- row-wise layernorm -> bf16 ------------------------
__global__ void k_ln_rows(const float* __restrict__ in,
                          __bf16* __restrict__ out, int rows) {
  int row = blockIdx.x * 8 + (threadIdx.x >> 5), lane = threadIdx.x & 31;
  if (row >= rows) return;
  const float* p = in + (size_t)row * Dm;
  float x[8];
  #pragma unroll
  for (int i = 0; i < 8; ++i) x[i] = p[lane + 32 * i];
  float s = 0.f, s2 = 0.f;
  #pragma unroll
  for (int i = 0; i < 8; ++i) { s += x[i]; s2 += x[i] * x[i]; }
  float m  = wave_sum(s) * (1.f / Dm);
  float vv = wave_sum(s2) * (1.f / Dm) - m * m;
  float r  = rsqrtf(vv + 1e-5f);
  #pragma unroll
  for (int i = 0; i < 8; ++i)
    out[(size_t)row * Dm + lane + 32 * i] = (__bf16)((x[i] - m) * r);
}

// -------- GEMM4: y = relu(ln(a) @ decoder_y[h]) * x ; TDM-staged A ---------
__global__ void k_gemm_y(const __bf16* __restrict__ ALn,  // H x 1024x256
                         const __bf16* __restrict__ BT,   // decyT H x 8192x256
                         const __bf16* __restrict__ X,    // H x 1024x8192
                         __bf16* __restrict__ Y) {        // 1024 x 32768
  __shared__ __bf16 sA[32 * LPITCH];
  int h = blockIdx.z, t0 = blockIdx.y * 32;
  int w = threadIdx.x >> 5, lane = threadIdx.x & 31;
  int n0 = blockIdx.x * 256 + w * 32;
  const __bf16* Ah = ALn + (size_t)h * Tt * Dm;
  if (threadIdx.x < 32)
    tdm_load_panel(Ah + (size_t)t0 * Dm, (unsigned)(uintptr_t)sA, Dm);
  __builtin_amdgcn_s_wait_tensorcnt(0);
  __syncthreads();
  const __bf16* Bh = BT + (size_t)h * NHd * Dm;
  f32x8 c00 = {}, c01 = {}, c10 = {}, c11 = {};
  #pragma unroll
  for (int k0 = 0; k0 < Dm; k0 += 32) {
    bh16 a0 = fragA_lds(sA, 0, k0),  a1 = fragA_lds(sA, 16, k0);
    bh16 b0 = load_fragB(Bh, Dm, n0, k0), b1 = load_fragB(Bh, Dm, n0 + 16, k0);
    c00 = wmma_bf16(a0, b0, c00); c01 = wmma_bf16(a0, b1, c01);
    c10 = wmma_bf16(a1, b0, c10); c11 = wmma_bf16(a1, b1, c11);
  }
  int col = n0 + (lane & 15), rb = (lane >> 4) * 8;
  const __bf16* Xh = X + (size_t)h * Tt * NHd;
  #pragma unroll
  for (int j = 0; j < 8; ++j) {
    int r = t0 + rb + j;
    float x00 = (float)Xh[(size_t)r * NHd + col];
    float x01 = (float)Xh[(size_t)r * NHd + col + 16];
    float x10 = (float)Xh[(size_t)(r + 16) * NHd + col];
    float x11 = (float)Xh[(size_t)(r + 16) * NHd + col + 16];
    Y[(size_t)r * Ntot + (size_t)h * NHd + col]             = (__bf16)(fmaxf(c00[j], 0.f) * x00);
    Y[(size_t)r * Ntot + (size_t)h * NHd + col + 16]        = (__bf16)(fmaxf(c01[j], 0.f) * x01);
    Y[(size_t)(r + 16) * Ntot + (size_t)h * NHd + col]      = (__bf16)(fmaxf(c10[j], 0.f) * x10);
    Y[(size_t)(r + 16) * Ntot + (size_t)h * NHd + col + 16] = (__bf16)(fmaxf(c11[j], 0.f) * x11);
  }
}

// ------- GEMM5: z_part[ks] = y @ encoder over K-chunk; TDM double-buffer ---
__global__ void k_gemm_enc(const __bf16* __restrict__ Y,   // 1024 x 32768
                           const __bf16* __restrict__ ET,  // 256 x 32768
                           float* __restrict__ Zp) {       // KS x 1024 x 256
  __shared__ __bf16 sA[2][32 * LPITCH];
  int t0 = blockIdx.y * 32, ks = blockIdx.z;
  int w = threadIdx.x >> 5, lane = threadIdx.x & 31;
  int n0 = w * 32;
  bool wave0 = (threadIdx.x < 32);
  const int Kc = Ntot / KS;                  // 4096
  const int P  = Kc / 256;                   // 16 panels
  const int kbase = ks * Kc;
  if (wave0)
    tdm_load_panel(Y + (size_t)t0 * Ntot + kbase,
                   (unsigned)(uintptr_t)sA[0], Ntot);
  f32x8 c00 = {}, c01 = {}, c10 = {}, c11 = {};
  for (int p = 0; p < P; ++p) {
    if (wave0 && p + 1 < P)
      tdm_load_panel(Y + (size_t)t0 * Ntot + kbase + (p + 1) * 256,
                     (unsigned)(uintptr_t)sA[(p + 1) & 1], Ntot);
    if (p + 1 < P) __builtin_amdgcn_s_wait_tensorcnt(1);
    else           __builtin_amdgcn_s_wait_tensorcnt(0);
    __syncthreads();
    const __bf16* sm = sA[p & 1];
    int kp = kbase + p * 256;
    #pragma unroll
    for (int k0 = 0; k0 < 256; k0 += 32) {
      __builtin_prefetch(ET + (size_t)(n0 + (lane & 15)) * Ntot + kp + k0 + 256, 0, 3);
      bh16 a0 = fragA_lds(sm, 0, k0),  a1 = fragA_lds(sm, 16, k0);
      bh16 b0 = load_fragB(ET, Ntot, n0, kp + k0);
      bh16 b1 = load_fragB(ET, Ntot, n0 + 16, kp + k0);
      c00 = wmma_bf16(a0, b0, c00); c01 = wmma_bf16(a0, b1, c01);
      c10 = wmma_bf16(a1, b0, c10); c11 = wmma_bf16(a1, b1, c11);
    }
    __syncthreads();
  }
  int col = n0 + (lane & 15), rb = (lane >> 4) * 8;
  float* Z = Zp + (size_t)ks * Tt * Dm;
  #pragma unroll
  for (int j = 0; j < 8; ++j) {
    int r = t0 + rb + j;
    Z[(size_t)r * Dm + col]             = c00[j];
    Z[(size_t)r * Dm + col + 16]        = c01[j];
    Z[(size_t)(r + 16) * Dm + col]      = c10[j];
    Z[(size_t)(r + 16) * Dm + col + 16] = c11[j];
  }
}

// ---------- v = ln(v + ln(sum_ks z_part)); refresh bf16 copies -------------
__global__ void k_update_v(const float* __restrict__ Zp,
                           float* __restrict__ v, __bf16* __restrict__ vb,
                           __bf16* __restrict__ vbT) {
  int row = blockIdx.x * 8 + (threadIdx.x >> 5), lane = threadIdx.x & 31;
  if (row >= Tt) return;
  float z[8];
  #pragma unroll
  for (int i = 0; i < 8; ++i) {
    float acc = 0.f;
    #pragma unroll
    for (int ks = 0; ks < KS; ++ks)
      acc += Zp[((size_t)ks * Tt + row) * Dm + lane + 32 * i];
    z[i] = acc;
  }
  float s = 0.f, s2 = 0.f;
  #pragma unroll
  for (int i = 0; i < 8; ++i) { s += z[i]; s2 += z[i] * z[i]; }
  float m  = wave_sum(s) * (1.f / Dm);
  float vv = wave_sum(s2) * (1.f / Dm) - m * m;
  float r  = rsqrtf(vv + 1e-5f);
  float wv[8];
  s = 0.f; s2 = 0.f;
  #pragma unroll
  for (int i = 0; i < 8; ++i) {
    wv[i] = v[(size_t)row * Dm + lane + 32 * i] + (z[i] - m) * r;
    s += wv[i]; s2 += wv[i] * wv[i];
  }
  m  = wave_sum(s) * (1.f / Dm);
  vv = wave_sum(s2) * (1.f / Dm) - m * m;
  r  = rsqrtf(vv + 1e-5f);
  #pragma unroll
  for (int i = 0; i < 8; ++i) {
    int c = lane + 32 * i;
    float y = (wv[i] - m) * r;
    v[(size_t)row * Dm + c]   = y;
    vb[(size_t)row * Dm + c]  = (__bf16)y;
    vbT[(size_t)c * Tt + row] = (__bf16)y;
  }
}

// ------------------------ out = v @ readout (f32 out) ----------------------
__global__ void k_gemm_out(const __bf16* __restrict__ A,
                           const __bf16* __restrict__ RT,
                           float* __restrict__ Out) {
  int t0 = blockIdx.y * 16;
  int n0 = (blockIdx.x * 8 + (threadIdx.x >> 5)) * 16;
  f32x8 acc = {};
  #pragma unroll
  for (int k0 = 0; k0 < Dm; k0 += 32)
    acc = wmma_bf16(load_fragA(A, Dm, t0, k0), load_fragB(RT, Dm, n0, k0), acc);
  int lane = threadIdx.x & 31, col = n0 + (lane & 15), rb = (lane >> 4) * 8;
  #pragma unroll
  for (int j = 0; j < 8; ++j)
    Out[(size_t)(t0 + rb + j) * Dm + col] = acc[j];
}

// ------------------------------ host side ----------------------------------
extern "C" void kernel_launch(void* const* d_in, const int* in_sizes, int n_in,
                              void* d_out, int out_size, void* d_ws,
                              size_t ws_size, hipStream_t stream) {
  (void)in_sizes; (void)n_in; (void)out_size; (void)ws_size;
  const int*   idx     = (const int*)d_in[0];
  const float* wte     = (const float*)d_in[1];
  const float* encoder = (const float*)d_in[2];
  const float* dec_x   = (const float*)d_in[3];
  const float* dec_y   = (const float*)d_in[4];
  const float* readout = (const float*)d_in[5];
  float* out = (float*)d_out;

  char* ws = (char*)d_ws;
  size_t off = 0;
  auto alloc = [&](size_t bytes) -> void* {
    void* p = ws + off;
    off = (off + bytes + 255) & ~(size_t)255;
    return p;
  };
  __bf16* decxT = (__bf16*)alloc((size_t)Hh * NHd * Dm * 2);
  __bf16* decyT = (__bf16*)alloc((size_t)Hh * NHd * Dm * 2);
  __bf16* encT  = (__bf16*)alloc((size_t)Dm * Ntot * 2);
  __bf16* routT = (__bf16*)alloc((size_t)Dm * VOC * 2);
  float*  v_f   = (float*)alloc((size_t)Tt * Dm * 4);
  __bf16* v_b   = (__bf16*)alloc((size_t)Tt * Dm * 2);
  __bf16* v_bT  = (__bf16*)alloc((size_t)Dm * Tt * 2);
  __bf16* Xb    = (__bf16*)alloc((size_t)Hh * Tt * NHd * 2);
  __bf16* XRb   = (__bf16*)alloc((size_t)Hh * Tt * NHd * 2);
  __bf16* Sb    = (__bf16*)alloc((size_t)Hh * Tt * Tt * 2);
  float*  Af    = (float*)alloc((size_t)Hh * Tt * Dm * 4);
  __bf16* ALn   = (__bf16*)alloc((size_t)Hh * Tt * Dm * 2);
  __bf16* Yb    = (__bf16*)alloc((size_t)Tt * Ntot * 2);
  float*  Zp    = (float*)alloc((size_t)KS * Tt * Dm * 4);

  dim3 blkT(32, 8);
  for (int h = 0; h < Hh; ++h) {
    k_cvt_T<<<dim3(NHd / 32, Dm / 32), blkT, 0, stream>>>(
        dec_x + (size_t)h * Dm * NHd, decxT + (size_t)h * NHd * Dm, Dm, NHd);
    k_cvt_T<<<dim3(NHd / 32, Dm / 32), blkT, 0, stream>>>(
        dec_y + (size_t)h * Dm * NHd, decyT + (size_t)h * NHd * Dm, Dm, NHd);
  }
  k_cvt_T<<<dim3(Dm / 32, Ntot / 32), blkT, 0, stream>>>(encoder, encT, Ntot, Dm);
  k_cvt_T<<<dim3(VOC / 32, Dm / 32), blkT, 0, stream>>>(readout, routT, Dm, VOC);

  k_embed<<<Tt / 8, 256, 0, stream>>>(idx, wte, v_f, v_b, v_bT);

  for (int l = 0; l < Lnum; ++l) {
    k_gemm_x<<<dim3(NHd / 256, Tt / 32, Hh), 256, 0, stream>>>(v_b, decxT, Xb);
    {
      long long total = (long long)Hh * Tt * HALF;
      k_rope<<<(unsigned)((total + 255) / 256), 256, 0, stream>>>(Xb, XRb);
    }
    k_gemm_scores<<<dim3(Tt / 256, Tt / 32, Hh), 256, 0, stream>>>(XRb, Sb);
    k_gemm_av<<<dim3(Dm / 128, Tt / 16, Hh), 256, 0, stream>>>(Sb, v_bT, Af);
    k_ln_rows<<<(Hh * Tt) / 8, 256, 0, stream>>>(Af, ALn, Hh * Tt);
    k_gemm_y<<<dim3(NHd / 256, Tt / 32, Hh), 256, 0, stream>>>(ALn, decyT, Xb, Yb);
    k_gemm_enc<<<dim3(1, Tt / 32, KS), 256, 0, stream>>>(Yb, encT, Zp);
    k_update_v<<<Tt / 8, 256, 0, stream>>>(Zp, v_f, v_b, v_bT);
  }
  k_gemm_out<<<dim3(VOC / 128, Tt / 16, 1), 256, 0, stream>>>(v_b, routT, out);
}
